// BoxAttention_33655363732242
// MI455X (gfx1250) — matrix operand order repaired and checked by reading, not compiled, not run on hardware
//
#include <hip/hip_runtime.h>
#include <hip/hip_bf16.h>
#include <stdint.h>

// ---------------------------------------------------------------------------
// Fused 3D-box window attention for MI455X (gfx1250, wave32, WMMA).
// One window (64 tokens x 384 ch) per 256-thread workgroup (8 wave32).
// All intermediates (x-f16, per-head q/k/vT, scores, attn, pre-proj out) live
// in LDS (~142 KB of the 320 KB/WGP). Every GEMM uses v_wmma_f32_16x16x32_f16.
// V is kept transposed in LDS so both WMMA operand streams are contiguous
// 128-bit LDS accesses (no 16-bit gathers).
// ---------------------------------------------------------------------------

typedef __attribute__((ext_vector_type(16))) _Float16 v16h;
typedef __attribute__((ext_vector_type(8)))  float    v8f;

#define N_TOK 64
#define DIM   384
#define NHEAD 12
#define HD    32
#define NWIN  512
#define SCALE 0.17677669529663687f   // 32^-0.5

#define XS_STRIDE 392   // 64x392 f16 (x tile / out tile), padded
#define QH_STRIDE 40    // 64x40  f16 per-head q/k, padded
#define VT_STRIDE 72    // 32x72  f16 per-head v^T (d-major), padded (144B rows)
#define SB_STRIDE 68    // 64x68  f32 score buffer, padded (272B rows, 16B align)
#define AT_STRIDE 72    // 64x72  f16 attn probs, padded

// LDS: xs 50176 | os 50176 | qh/kh 2*5120 | vt 4608 | sbuf 17408 | attn 9216
static constexpr size_t SMEM_BYTES =
    (size_t)(64 * XS_STRIDE) * 2 * 2 +      // xs + os
    (size_t)(64 * QH_STRIDE) * 2 * 2 +      // qh, kh
    (size_t)(32 * VT_STRIDE) * 2 +          // vt
    (size_t)(64 * SB_STRIDE) * 4 +          // sbuf (f32)
    (size_t)(64 * AT_STRIDE) * 2;           // attn (f16)  == 141824

union AFrag { v16h v; uint32_t u[8]; };
union BFrag { v16h v; uint4 q[2]; };
union H8    { uint4 q; _Float16 h[8]; };
union H4    { uint2 q; _Float16 h[4]; };

// A-matrix 16x32 f16 fragment per ISA 7.12.2: lane = (m, hi); VGPR j<4 holds
// K={2j,2j+1}+8*hi, VGPR j>=4 holds K={16+2(j-4),...}+8*hi.
// rowptr = &A[row][k0]  (4-byte aligned)
__device__ __forceinline__ v16h load_a_frag(const _Float16* rowptr, int hi) {
  AFrag f;
  const uint32_t* p = (const uint32_t*)rowptr;
  const int o = 4 * hi;                    // 8 halfs = 4 dwords
  f.u[0] = p[0 + o];  f.u[1] = p[1 + o];
  f.u[2] = p[2 + o];  f.u[3] = p[3 + o];
  f.u[4] = p[8 + o];  f.u[5] = p[9 + o];
  f.u[6] = p[10 + o]; f.u[7] = p[11 + o];
  return f.v;
}

// B-matrix 32x16 f16 fragment, K contiguous in memory: lane = (n, hi),
// element e<8 -> K = e + 8*hi, e>=8 -> K = 16 + 8*hi + (e-8). p = &Bstore[n][k0] where Bstore[n][k] = B[k][n].
// Two 16B loads (ds_load_b128 / global_load_b128).
__device__ __forceinline__ v16h load_b_contig(const _Float16* p, int hi) {
  BFrag f;
  const uint4* q = (const uint4*)(p + 8 * hi);   // 16B aligned
  f.q[0] = q[0];
  f.q[1] = q[2];
  return f.v;
}

__device__ __forceinline__ v8f wmma_f16(v16h a, v16h b, v8f c) {
  return __builtin_amdgcn_wmma_f32_16x16x32_f16(false, a, false, b,
                                                (short)0, c, false, false);
}

// ---------------------------------------------------------------------------
// Kernel 0: one-time setup. f32 -> f16 weights + expand relative-position
// bias table into a dense [H][N][N] f32 matrix (shared by all 2048 windows,
// L2-resident).
// ---------------------------------------------------------------------------
extern "C" __global__ __launch_bounds__(256)
void boxattn_setup(const float* __restrict__ wqkv,
                   const float* __restrict__ wproj,
                   const float* __restrict__ bias_table,
                   _Float16* __restrict__ wqkv_h,
                   _Float16* __restrict__ wproj_h,
                   float* __restrict__ bexp) {
  const int NQ = 3 * DIM * DIM;             // 442368
  const int NP = DIM * DIM;                 // 147456
  const int NB = NHEAD * N_TOK * N_TOK;     // 49152
  int i = blockIdx.x * 256 + threadIdx.x;
  if (i < NQ) {
    wqkv_h[i] = (_Float16)wqkv[i];
  } else if (i < NQ + NP) {
    wproj_h[i - NQ] = (_Float16)wproj[i - NQ];
  } else if (i < NQ + NP + NB) {
    int j = i - NQ - NP;
    int h = j / (N_TOK * N_TOK);
    int r = j % (N_TOK * N_TOK);
    int n = r >> 6, m = r & 63;
    int zn = n >> 4, yn = (n >> 2) & 3, xn = n & 3;
    int zm = m >> 4, ym = (m >> 2) & 3, xm = m & 3;
    int idx = (zn - zm + 3) * 49 + (yn - ym + 3) * 7 + (xn - xm + 3);
    bexp[j] = bias_table[idx * NHEAD + h];
  }
}

// ---------------------------------------------------------------------------
// Kernel 1: fused per-window attention.
// ---------------------------------------------------------------------------
extern "C" __global__ __launch_bounds__(256)
void boxattn_fused(const float* __restrict__ x,
                   const float* __restrict__ mask,
                   const _Float16* __restrict__ wqkv,
                   const float* __restrict__ bexp,
                   const _Float16* __restrict__ wproj,
                   const float* __restrict__ bproj,
                   float* __restrict__ out) {
  extern __shared__ char smem[];
  _Float16* xs   = (_Float16*)smem;                 // [64][392] f16 (x tile)
  _Float16* os   = xs + 64 * XS_STRIDE;             // [64][392] f16 (pre-proj out)
  _Float16* qh   = os + 64 * XS_STRIDE;             // [64][40] f16 (per-head q)
  _Float16* kh   = qh + 64 * QH_STRIDE;             // [64][40]
  _Float16* vt   = kh + 64 * QH_STRIDE;             // [32][72] f16 (v^T, d-major)
  float*    sbuf = (float*)(vt + 32 * VT_STRIDE);   // [64][68] f32 scores
  _Float16* attn = (_Float16*)(sbuf + 64 * SB_STRIDE); // [64][72] f16 probs

  const int b    = blockIdx.x;
  const int tid  = threadIdx.x;
  const int wave = tid >> 5;
  const int lane = tid & 31;
  const int lm   = lane & 15;   // n / m position within 16
  const int hi   = lane >> 4;   // half-wave select

  // ---- Stage A: load x[b] (64x384 f32), convert to f16 in LDS -------------
  {
    const float4* xg = (const float4*)(x + (size_t)b * (N_TOK * DIM));
    for (int i = tid; i < (N_TOK * DIM) / 4; i += 256) {
      float4 f = xg[i];
      int row = (i * 4) / DIM, col = (i * 4) % DIM;
      _Float16* d = xs + row * XS_STRIDE + col;
      d[0] = (_Float16)f.x; d[1] = (_Float16)f.y;
      d[2] = (_Float16)f.z; d[3] = (_Float16)f.w;
    }
  }
  __syncthreads();

  // ---- Per-head attention --------------------------------------------------
  for (int h = 0; h < NHEAD; ++h) {
    // Stage B: per-head qkv GEMM. 24 output tiles (4 row-tiles x 6 col-tiles
    // spanning q|k|v blocks of 32 cols each); each wave owns 3 tiles in one
    // row so the A fragment is shared. K loop: 384/32 = 12 WMMAs per tile.
    // q,k stored token-major; v stored transposed (d-major) with one packed
    // ds_store_b128 per lane per tile.
    {
      const int mt    = wave >> 1;
      const int cbase = (wave & 1) * 3;
      v8f acc[3] = {};
      for (int kk = 0; kk < DIM; kk += 32) {
        v16h a = load_a_frag(xs + (mt * 16 + lm) * XS_STRIDE + kk, hi);
#pragma unroll
        for (int j = 0; j < 3; ++j) {
          int ct  = cbase + j;                       // 0..5
          int blk = ct >> 1;                         // 0=q 1=k 2=v
          int o   = blk * DIM + h * HD + (ct & 1) * 16 + lm;   // wqkv row
          v16h bf = load_b_contig(wqkv + (size_t)o * DIM + kk, hi);
          acc[j] = wmma_f16(a, bf, acc[j]);
        }
      }
#pragma unroll
      for (int j = 0; j < 3; ++j) {
        int ct   = cbase + j;
        int blk  = ct >> 1;
        int coff = (ct & 1) * 16 + lm;               // channel within head
        if (blk == 2) {
          // v^T: lane's 8 values are 8 consecutive tokens of channel coff.
          H8 pk;
#pragma unroll
          for (int r = 0; r < 8; ++r) pk.h[r] = (_Float16)acc[j][r];
          *(uint4*)(vt + coff * VT_STRIDE + mt * 16 + 8 * hi) = pk.q;
        } else {
          _Float16* dst = (blk == 0) ? qh : kh;
          float sc = (blk == 0) ? SCALE : 1.0f;      // fold softmax scale into q
#pragma unroll
          for (int r = 0; r < 8; ++r)
            dst[(mt * 16 + r + 8 * hi) * QH_STRIDE + coff] = (_Float16)(acc[j][r] * sc);
        }
      }
    }
    __syncthreads();

    // Stage C1: S = q @ k^T. 16 tiles of 16x16, K = HD = 32 -> one WMMA each.
    // Both tiles of a wave share the row tile -> single A fragment.
    {
      const int mt = wave >> 1;
      v16h a = load_a_frag(qh + (mt * 16 + lm) * QH_STRIDE, hi);
#pragma unroll
      for (int t = 0; t < 2; ++t) {
        int nt = (wave & 1) * 2 + t;
        v16h bf = load_b_contig(kh + (nt * 16 + lm) * QH_STRIDE, hi);
        v8f c = {};
        c = wmma_f16(a, bf, c);
#pragma unroll
        for (int r = 0; r < 8; ++r)
          sbuf[(mt * 16 + r + 8 * hi) * SB_STRIDE + nt * 16 + lm] = c[r];
      }
    }
    __syncthreads();

    // Stage C2: softmax rows + precomputed bias + window mask (all float4
    // streams). One thread per token row (64 active); online max/sum.
    if (tid < N_TOK) {
      const int n = tid;
      const float4* m4 = (const float4*)(mask +
          (size_t)(b & (NWIN - 1)) * (N_TOK * N_TOK) + n * N_TOK);
      const float4* b4 = (const float4*)(bexp +
          (size_t)h * (N_TOK * N_TOK) + n * N_TOK);
      float4* s4 = (float4*)(sbuf + n * SB_STRIDE);
      float mx = -1e30f, sum = 0.0f;
#pragma unroll 4
      for (int i = 0; i < N_TOK / 4; ++i) {
        float4 sv = s4[i], bv = b4[i], mv = m4[i];
        sv.x += bv.x + mv.x; sv.y += bv.y + mv.y;
        sv.z += bv.z + mv.z; sv.w += bv.w + mv.w;
        s4[i] = sv;
        float lm4 = fmaxf(fmaxf(sv.x, sv.y), fmaxf(sv.z, sv.w));
        float mo = mx;
        mx = fmaxf(mx, lm4);
        sum = sum * __expf(mo - mx) +
              __expf(sv.x - mx) + __expf(sv.y - mx) +
              __expf(sv.z - mx) + __expf(sv.w - mx);
      }
      float inv = 1.0f / sum;
#pragma unroll 4
      for (int i = 0; i < N_TOK / 4; ++i) {
        float4 sv = s4[i];
        H4 pk;
        pk.h[0] = (_Float16)(__expf(sv.x - mx) * inv);
        pk.h[1] = (_Float16)(__expf(sv.y - mx) * inv);
        pk.h[2] = (_Float16)(__expf(sv.z - mx) * inv);
        pk.h[3] = (_Float16)(__expf(sv.w - mx) * inv);
        *(uint2*)(attn + n * AT_STRIDE + i * 4) = pk.q;
      }
    }
    __syncthreads();

    // Stage C3: O_h = P @ v. 8 tiles (4x2), one per wave, K = 64 -> 2 WMMAs.
    // v^T rows are contiguous in K (tokens) -> ds_load_b128 B fragments.
    {
      const int mt = wave >> 1, nt = wave & 1;
      v8f c = {};
#pragma unroll
      for (int kk = 0; kk < N_TOK; kk += 32) {
        v16h a  = load_a_frag(attn + (mt * 16 + lm) * AT_STRIDE + kk, hi);
        v16h bf = load_b_contig(vt + (nt * 16 + lm) * VT_STRIDE + kk, hi);
        c = wmma_f16(a, bf, c);
      }
#pragma unroll
      for (int r = 0; r < 8; ++r)
        os[(mt * 16 + r + 8 * hi) * XS_STRIDE + h * HD + nt * 16 + lm] = (_Float16)c[r];
    }
    __syncthreads();   // also guards qh/kh/vt overwrite next head
  }

  // ---- Stage D: out = os @ wproj^T + bproj, f32 store ----------------------
  {
    const int mt    = wave >> 1;      // row tile
    const int halfc = wave & 1;       // 12 col-tiles per wave, 6 at a time
    float* og = out + (size_t)b * (N_TOK * DIM);
    for (int cchunk = 0; cchunk < 2; ++cchunk) {
      v8f acc[6] = {};
      const int ntbase = halfc * 12 + cchunk * 6;
      for (int kk = 0; kk < DIM; kk += 32) {
        v16h a = load_a_frag(os + (mt * 16 + lm) * XS_STRIDE + kk, hi);
#pragma unroll
        for (int j = 0; j < 6; ++j) {
          int o = (ntbase + j) * 16 + lm;
          v16h bf = load_b_contig(wproj + (size_t)o * DIM + kk, hi);
          acc[j] = wmma_f16(a, bf, acc[j]);
        }
      }
#pragma unroll
      for (int j = 0; j < 6; ++j) {
        int col = (ntbase + j) * 16 + lm;
        float bp = bproj[col];
#pragma unroll
        for (int r = 0; r < 8; ++r)
          og[(size_t)(mt * 16 + r + 8 * hi) * DIM + col] = acc[j][r] + bp;
      }
    }
  }
}

// ---------------------------------------------------------------------------
extern "C" void kernel_launch(void* const* d_in, const int* in_sizes, int n_in,
                              void* d_out, int out_size, void* d_ws, size_t ws_size,
                              hipStream_t stream) {
  const float* x          = (const float*)d_in[0];
  const float* mask       = (const float*)d_in[1];
  const float* w_qkv      = (const float*)d_in[2];
  const float* bias_table = (const float*)d_in[3];
  const float* w_proj     = (const float*)d_in[4];
  const float* b_proj     = (const float*)d_in[5];
  float* out = (float*)d_out;

  // Workspace: f16 weights + expanded bias.
  // [0, 884736)          : wqkv_h  (3*384*384 f16)
  // [884736, 1179648)    : wproj_h (384*384 f16)
  // [1179648, 1376256)   : bexp    (12*64*64 f32)
  _Float16* wqkv_h  = (_Float16*)d_ws;
  _Float16* wproj_h = wqkv_h + 3 * DIM * DIM;
  float*    bexp    = (float*)((char*)d_ws + 1179648);

  const int total = 3 * DIM * DIM + DIM * DIM + NHEAD * N_TOK * N_TOK;
  boxattn_setup<<<(total + 255) / 256, 256, 0, stream>>>(
      w_qkv, w_proj, bias_table, wqkv_h, wproj_h, bexp);

  boxattn_fused<<<2048, 256, SMEM_BYTES, stream>>>(
      x, mask, wqkv_h, bexp, wproj_h, b_proj, out);
}
